// WolfPQ_78520592106009
// MI455X (gfx1250) — compile-verified, matrix operand order checked by reading, not run on hardware
//
#include <hip/hip_runtime.h>
#include <math.h>

// ---------------------------------------------------------------------------
// WolfPQ forward, CDNA5 (gfx1250, wave32, WMMA bf16 16x16x32)
//
//   h = tanh(x @ W1 + b1)                      GEMM1 (bf16 WMMA)
//   a = relu(h @ W2 + b2)                      GEMM2 (bf16 WMMA)
//   idx = argmax_k(a.reshape(B,M,K) + gumbel)  fused epilogue (no exp needed:
//   s   = one_hot(idx)                          s == y_hard exactly, and
//   res2[b, m*16:] = codebook[m, idx[b,m], :]   argmax(softmax(z))==argmax(z))
//
// Staging uses CDNA5 GLOBAL_LOAD_ASYNC_TO_LDS (ASYNCcnt) with double-buffered
// LDS when the builtins are available; otherwise falls back to VGPR staging.
// ---------------------------------------------------------------------------

#define BATCH 16384
#define DIM   768
#define MCB   48
#define KCB   256
#define MK2   6144      // M*K/2
#define MK    12288     // M*K
#define SUB   16        // DIM/M

typedef __bf16 bf16_t;
typedef bf16_t v16bf __attribute__((ext_vector_type(16)));
typedef bf16_t v8bf  __attribute__((ext_vector_type(8)));
typedef float  v8f   __attribute__((ext_vector_type(8)));

#if defined(__has_builtin)
#if __has_builtin(__builtin_amdgcn_global_load_async_to_lds_b128) && \
    __has_builtin(__builtin_amdgcn_s_wait_asynccnt)
#define USE_ASYNC_LDS 1
#endif
#endif

#ifdef USE_ASYNC_LDS
// Builtin signature (from clang diagnostics): (int4 AS1*, int4 AS3*, Imm, Imm).
// Route casts through integers: AS1 shares the generic 64-bit representation;
// generic->LDS is addr[31:0] per ISA 10.2, so a 32-bit truncation is the
// correct AS3 offset.
typedef int i32x4 __attribute__((ext_vector_type(4)));
typedef __attribute__((address_space(1))) i32x4 as1_i32x4;
typedef __attribute__((address_space(3))) i32x4 as3_i32x4;
#define ASYNC_CP_B128(gp, lp)                                            \
    __builtin_amdgcn_global_load_async_to_lds_b128(                      \
        (as1_i32x4*)(unsigned long long)(gp),                            \
        (as3_i32x4*)(unsigned int)(unsigned long long)(lp), 0, 0)
#define WOLF_WAIT_ASYNC(n) __builtin_amdgcn_s_wait_asynccnt(n)
#else
#define WOLF_WAIT_ASYNC(n)
#endif

__device__ __forceinline__ unsigned short f2bf(float f) {
    unsigned u = __builtin_bit_cast(unsigned, f);
    u += 0x7FFFu + ((u >> 16) & 1u);   // round-to-nearest-even
    return (unsigned short)(u >> 16);
}

// Load a 16x32 bf16 WMMA A/B fragment from a row-major LDS tile with row
// stride 40 bf16 (80B, padded to avoid bank conflicts).  Per the CDNA5 ISA
// 16-bit A-matrix layout: lane L<16 holds row L, K in {0..7}u{16..23};
// lane L+16 holds row L, K in {8..15}u{24..31}.  B uses the same pattern with
// "row" = N column (weights are stored transposed [N][K]).
__device__ __forceinline__ v16bf load_frag(const bf16_t* __restrict__ p0, int lane) {
    const bf16_t* p = p0 + (lane & 15) * 40 + ((lane >> 4) << 3);
    union { v16bf v; v8bf h[2]; } u;
    u.h[0] = *(const v8bf*)(p);
    u.h[1] = *(const v8bf*)(p + 16);
    return u.v;
}

// ---------------------------------------------------------------------------
// elementwise fp32 -> bf16 (x)
// ---------------------------------------------------------------------------
__global__ __launch_bounds__(256)
void wolfpq_cvt_bf16(const float* __restrict__ in, unsigned short* __restrict__ out, int n) {
    int i = (blockIdx.x * 256 + threadIdx.x) * 8;
    if (i + 8 > n) return;
    float4 a = *(const float4*)(in + i);
    float4 b = *(const float4*)(in + i + 4);
    uint4 o;
    o.x = (unsigned)f2bf(a.x) | ((unsigned)f2bf(a.y) << 16);
    o.y = (unsigned)f2bf(a.z) | ((unsigned)f2bf(a.w) << 16);
    o.z = (unsigned)f2bf(b.x) | ((unsigned)f2bf(b.y) << 16);
    o.w = (unsigned)f2bf(b.z) | ((unsigned)f2bf(b.w) << 16);
    *(uint4*)(out + i) = o;
}

// ---------------------------------------------------------------------------
// fp32 [R][C] -> bf16 [C][R]   (weights to N-major for WMMA B fragments)
// ---------------------------------------------------------------------------
__global__ __launch_bounds__(256)
void wolfpq_transpose_bf16(const float* __restrict__ in, unsigned short* __restrict__ out,
                           int R, int C) {
    __shared__ __align__(16) unsigned short tile[32][33];
    int cx = blockIdx.x * 32 + threadIdx.x;            // input col
    #pragma unroll
    for (int j = 0; j < 4; ++j) {
        int ry = blockIdx.y * 32 + threadIdx.y + j * 8; // input row
        tile[threadIdx.y + j * 8][threadIdx.x] = f2bf(in[(size_t)ry * C + cx]);
    }
    __syncthreads();
    int rx = blockIdx.y * 32 + threadIdx.x;            // output col (= input row)
    #pragma unroll
    for (int j = 0; j < 4; ++j) {
        int cy = blockIdx.x * 32 + threadIdx.y + j * 8; // output row (= input col)
        out[(size_t)cy * R + rx] = tile[threadIdx.x][threadIdx.y + j * 8];
    }
}

// ---------------------------------------------------------------------------
// GEMM1: h = tanh(x @ W1 + b1), bf16 in/out.  Tile 128x128, K=768.
// 8 waves: wave w owns rows [w*16, w*16+16), 8 col-tiles -> 8 accumulators.
// ---------------------------------------------------------------------------
__global__ __launch_bounds__(256)
void wolfpq_gemm1_tanh(const bf16_t* __restrict__ xb,      // [B][768]
                       const bf16_t* __restrict__ w1t,     // [6144][768]
                       const float*  __restrict__ b1,      // [6144]
                       unsigned short* __restrict__ hout)  // bf16 [B][6144]
{
    const int b0  = blockIdx.x * 128;
    const int n0  = blockIdx.y * 128;
    const int tid = threadIdx.x;
    const int wave = tid >> 5;
    const int lane = tid & 31;

    __shared__ __align__(16) bf16_t As[2][128 * 40];
    __shared__ __align__(16) bf16_t Bs[2][128 * 40];

    v8f acc[8];
    #pragma unroll
    for (int j = 0; j < 8; ++j)
        #pragma unroll
        for (int i = 0; i < 8; ++i) acc[j][i] = 0.0f;

    // 4 async (or staged) 16B transfers per thread per stage (2 A + 2 B)
    auto stage = [&](int buf, int k0) {
        #pragma unroll
        for (int i = 0; i < 2; ++i) {
            int c = tid + i * 256, r = c >> 2, kc = (c & 3) << 3;
#ifdef USE_ASYNC_LDS
            ASYNC_CP_B128(xb  + (size_t)(b0 + r) * DIM + k0 + kc, &As[buf][r * 40 + kc]);
            ASYNC_CP_B128(w1t + (size_t)(n0 + r) * DIM + k0 + kc, &Bs[buf][r * 40 + kc]);
#else
            uint4 va = *(const uint4*)(xb  + (size_t)(b0 + r) * DIM + k0 + kc);
            uint4 vb = *(const uint4*)(w1t + (size_t)(n0 + r) * DIM + k0 + kc);
            *(uint4*)(&As[buf][r * 40 + kc]) = va;
            *(uint4*)(&Bs[buf][r * 40 + kc]) = vb;
#endif
        }
    };

    auto compute = [&](int buf) {
        v16bf a = load_frag(&As[buf][wave * 16 * 40], lane);
        #pragma unroll
        for (int j = 0; j < 8; ++j) {
            v16bf bm = load_frag(&Bs[buf][j * 16 * 40], lane);
            acc[j] = __builtin_amdgcn_wmma_f32_16x16x32_bf16(
                false, a, false, bm, (short)0, acc[j], false, false);
        }
    };

    const int NK = DIM / 32;   // 24
    stage(0, 0);
    for (int kk = 0; kk < NK - 1; ++kk) {
        int cur = kk & 1;
        stage(cur ^ 1, (kk + 1) * 32);   // stage N+1 overlaps wait/compute on N
        WOLF_WAIT_ASYNC(4);              // stage N done; N+1 may be in flight
        __syncthreads();
        compute(cur);
        __syncthreads();                 // all waves done reading buf[cur]
    }
    WOLF_WAIT_ASYNC(0);
    __syncthreads();
    compute((NK - 1) & 1);

    // epilogue: tanh + bf16 store.  C layout: vgpr i, lanes 0-15 -> M=i,
    // lanes 16-31 -> M=i+8; N = lane&15 within col-tile j.
    const int halflane = lane & 15;
    const int rowhi    = (lane >> 4) << 3;
    #pragma unroll
    for (int i = 0; i < 8; ++i) {
        size_t grow = (size_t)(b0 + wave * 16 + i + rowhi);
        #pragma unroll
        for (int j = 0; j < 8; ++j) {
            int col = n0 + j * 16 + halflane;
            float v = tanhf(acc[j][i] + b1[col]);
            hout[grow * MK2 + col] = f2bf(v);
        }
    }
}

// ---------------------------------------------------------------------------
// GEMM2 + fused epilogue.  Tile 128 rows x 256 cols (one full m-group),
// K=6144.  8 waves: wave w owns rows [w*16, ..), all 16 col-tiles.
// Epilogue: val = relu(acc+b2) + gumbel; argmax over K=256; write one-hot s
// and gather codebook row into res2.  (s == y_hard exactly; no exp needed.)
// ---------------------------------------------------------------------------
__global__ __launch_bounds__(256)
void wolfpq_gemm2_argmax(const bf16_t* __restrict__ hmat,   // [B][6144]
                         const bf16_t* __restrict__ w2t,    // [12288][6144]
                         const float*  __restrict__ b2,     // [12288]
                         const float*  __restrict__ gumbel, // [B][12288]
                         const float*  __restrict__ codebook,// [48][256][16]
                         float* __restrict__ res2,          // [B][768]
                         float* __restrict__ s_out)         // [B][12288]
{
    const int b0  = blockIdx.x * 128;
    const int m   = blockIdx.y;           // 0..47
    const int n0  = m * KCB;
    const int tid = threadIdx.x;
    const int wave = tid >> 5;
    const int lane = tid & 31;

    __shared__ __align__(16) bf16_t As[2][128 * 40];
    __shared__ __align__(16) bf16_t Bs[2][256 * 40];
    __shared__ float candv[128 * 16];
    __shared__ int   candi[128 * 16];
    __shared__ int   rowidx[128];

    v8f acc[16];
    #pragma unroll
    for (int j = 0; j < 16; ++j)
        #pragma unroll
        for (int i = 0; i < 8; ++i) acc[j][i] = 0.0f;

    // 6 async (or staged) 16B transfers per thread per stage (2 A + 4 B)
    auto stage = [&](int buf, int k0) {
        #pragma unroll
        for (int i = 0; i < 2; ++i) {
            int c = tid + i * 256, r = c >> 2, kc = (c & 3) << 3;
#ifdef USE_ASYNC_LDS
            ASYNC_CP_B128(hmat + (size_t)(b0 + r) * MK2 + k0 + kc, &As[buf][r * 40 + kc]);
#else
            uint4 v = *(const uint4*)(hmat + (size_t)(b0 + r) * MK2 + k0 + kc);
            *(uint4*)(&As[buf][r * 40 + kc]) = v;
#endif
        }
        #pragma unroll
        for (int i = 0; i < 4; ++i) {
            int c = tid + i * 256, r = c >> 2, kc = (c & 3) << 3;
#ifdef USE_ASYNC_LDS
            ASYNC_CP_B128(w2t + (size_t)(n0 + r) * MK2 + k0 + kc, &Bs[buf][r * 40 + kc]);
#else
            uint4 v = *(const uint4*)(w2t + (size_t)(n0 + r) * MK2 + k0 + kc);
            *(uint4*)(&Bs[buf][r * 40 + kc]) = v;
#endif
        }
    };

    auto compute = [&](int buf) {
        v16bf a = load_frag(&As[buf][wave * 16 * 40], lane);
        #pragma unroll
        for (int j = 0; j < 16; ++j) {
            v16bf bm = load_frag(&Bs[buf][j * 16 * 40], lane);
            acc[j] = __builtin_amdgcn_wmma_f32_16x16x32_bf16(
                false, a, false, bm, (short)0, acc[j], false, false);
        }
    };

    const int NK = MK2 / 32;   // 192
    stage(0, 0);
    for (int kk = 0; kk < NK - 1; ++kk) {
        int cur = kk & 1;
        stage(cur ^ 1, (kk + 1) * 32);   // stage N+1 overlaps wait/compute on N
        WOLF_WAIT_ASYNC(6);              // stage N done; N+1 may be in flight
        __syncthreads();
        compute(cur);
        __syncthreads();                 // all waves done reading buf[cur]
    }
    WOLF_WAIT_ASYNC(0);
    __syncthreads();
    compute((NK - 1) & 1);

    // ---- per-lane argmax over col-tiles ----
    const int halflane = lane & 15;
    const int rowhi    = (lane >> 4) << 3;
    float b2v[16];
    #pragma unroll
    for (int j = 0; j < 16; ++j) b2v[j] = b2[n0 + j * 16 + halflane];

    #pragma unroll
    for (int i = 0; i < 8; ++i) {
        int rloc = wave * 16 + i + rowhi;
        size_t gbase = (size_t)(b0 + rloc) * MK + n0;
        float best = -3.4e38f; int bestc = 0;
        #pragma unroll
        for (int j = 0; j < 16; ++j) {
            int col = j * 16 + halflane;
            float v = acc[j][i] + b2v[j];
            v = v > 0.0f ? v : 0.0f;              // relu
            v += gumbel[gbase + col];
            if (v > best) { best = v; bestc = col; }   // cols ascend per lane
        }
        candv[rloc * 16 + halflane] = best;
        candi[rloc * 16 + halflane] = bestc;
    }
    __syncthreads();

    // ---- cross-lane reduction: 16 candidates per row ----
    if (tid < 128) {
        float best = candv[tid * 16]; int bi = candi[tid * 16];
        #pragma unroll
        for (int t = 1; t < 16; ++t) {
            float v = candv[tid * 16 + t];
            int   c = candi[tid * 16 + t];
            if (v > best || (v == best && c < bi)) { best = v; bi = c; }  // first-occurrence tie-break
        }
        rowidx[tid] = bi;
    }
    __syncthreads();

    // ---- write s = one_hot(idx): 128 rows x 256 cols ----
    {
        int r  = tid >> 1;
        int ch = (tid & 1) * 128;
        int idx = rowidx[r];
        size_t base = (size_t)(b0 + r) * MK + n0 + ch;
        #pragma unroll 4
        for (int c = 0; c < 128; c += 4) {
            float4 v = make_float4(0.f, 0.f, 0.f, 0.f);
            int cc = ch + c;
            if (idx >= cc && idx < cc + 4) ((float*)&v)[idx - cc] = 1.0f;
            *(float4*)(s_out + base + c) = v;
        }
    }

    // ---- res2[b, m*16 : m*16+16] = codebook[m, idx, :] ----
    {
        int r    = tid >> 1;
        int part = (tid & 1) * 8;
        int idx  = rowidx[r];
        const float* cb = codebook + ((size_t)m * KCB + idx) * SUB + part;
        float4 v0 = *(const float4*)cb;
        float4 v1 = *(const float4*)(cb + 4);
        float* dst = res2 + (size_t)(b0 + r) * DIM + m * SUB + part;
        *(float4*)dst       = v0;
        *(float4*)(dst + 4) = v1;
    }
}

// ---------------------------------------------------------------------------
extern "C" void kernel_launch(void* const* d_in, const int* in_sizes, int n_in,
                              void* d_out, int out_size, void* d_ws, size_t ws_size,
                              hipStream_t stream) {
    (void)in_sizes; (void)n_in; (void)out_size;
    const float* x        = (const float*)d_in[0];   // [B,768]
    const float* W1       = (const float*)d_in[1];   // [768,6144]
    const float* b1       = (const float*)d_in[2];   // [6144]
    const float* W2       = (const float*)d_in[3];   // [6144,12288]
    const float* b2       = (const float*)d_in[4];   // [12288]
    const float* codebook = (const float*)d_in[5];   // [48,256,16]
    const float* gumbel   = (const float*)d_in[6];   // [B,48,256]

    float* res2  = (float*)d_out;                        // [B,768]
    float* s_out = res2 + (size_t)BATCH * DIM;           // [B,12288]

    char* ws = (char*)d_ws;
    size_t off = 0;
    bf16_t* xb  = (bf16_t*)(ws + off); off += (size_t)BATCH * DIM * 2;   // 25.2 MB
    bf16_t* w1t = (bf16_t*)(ws + off); off += (size_t)MK2 * DIM * 2;     //  9.4 MB
    bf16_t* w2t = (bf16_t*)(ws + off); off += (size_t)MK * MK2 * 2;      // 151 MB
    bf16_t* hm  = (bf16_t*)(ws + off); off += (size_t)BATCH * MK2 * 2;   // 201 MB
    if (ws_size < off) return;

    // 1) x -> bf16
    wolfpq_cvt_bf16<<<(BATCH * DIM) / (256 * 8), 256, 0, stream>>>(
        x, (unsigned short*)xb, BATCH * DIM);
    // 2) W1 [768,6144] -> W1t bf16 [6144,768]
    wolfpq_transpose_bf16<<<dim3(MK2 / 32, DIM / 32), dim3(32, 8), 0, stream>>>(
        W1, (unsigned short*)w1t, DIM, MK2);
    // 3) W2 [6144,12288] -> W2t bf16 [12288,6144]
    wolfpq_transpose_bf16<<<dim3(MK / 32, MK2 / 32), dim3(32, 8), 0, stream>>>(
        W2, (unsigned short*)w2t, MK2, MK);
    // 4) h = tanh(x@W1 + b1)
    wolfpq_gemm1_tanh<<<dim3(BATCH / 128, MK2 / 128), 256, 0, stream>>>(
        xb, w1t, b1, (unsigned short*)hm);
    // 5) a = relu(h@W2 + b2); idx = argmax(a+gumbel); s = onehot; res2 = gather
    wolfpq_gemm2_argmax<<<dim3(BATCH / 128, MCB), 256, 0, stream>>>(
        hm, w2t, b2, gumbel, codebook, res2, s_out);
}
